// HyperConv2d_86758339379726
// MI455X (gfx1250) — matrix operand order repaired and checked by reading, not compiled, not run on hardware
//
#include <hip/hip_runtime.h>

typedef __attribute__((ext_vector_type(16))) _Float16 v16h;
typedef __attribute__((ext_vector_type(4)))  _Float16 v4h;
typedef __attribute__((ext_vector_type(8)))  float    v8f;

#define B_      16
#define INC_    128
#define OUTC_   256
#define ZDIM_   256
#define HW_     64
#define FANIN_  1152           // 128*9
#define NOUT_   4096           // 64*64
#define OTOT_   294912         // OUTC_*FANIN_
#define KSTEPS_ 36             // 9 taps * (128/32) ic-blocks

// ---------------- kernel 0: x f32 -> f16 ----------------
__global__ __launch_bounds__(256) void cvt_x_kernel(const float* __restrict__ x,
                                                    _Float16* __restrict__ xh) {
  size_t i = ((size_t)blockIdx.x * 256 + threadIdx.x) * 4;
  float4 v = *(const float4*)(x + i);
  v4h h = { (_Float16)v.x, (_Float16)v.y, (_Float16)v.z, (_Float16)v.w };
  *(v4h*)(xh + i) = h;
}

// ---------------- kernel 1: delta = z @ head_w^T  (memory-bound, f32) ----------
// head_w: [OTOT_, ZDIM_] row-major; out delta: [OTOT_][16] (contiguous 64B per o)
__global__ __launch_bounds__(256) void delta_gemm_kernel(const float* __restrict__ z,
                                                         const float* __restrict__ hw,
                                                         float* __restrict__ delta) {
  __shared__ float zl[B_ * ZDIM_];          // 16 KB
  for (int i = threadIdx.x; i < B_ * ZDIM_; i += 256) zl[i] = z[i];
  __syncthreads();

  const size_t o = (size_t)blockIdx.x * 256 + threadIdx.x;
  float acc[B_];
#pragma unroll
  for (int b = 0; b < B_; ++b) acc[b] = 0.f;

  const float* row = hw + o * ZDIM_;
  for (int j = 0; j < ZDIM_; j += 4) {
    float4 h = *(const float4*)(row + j);
#pragma unroll
    for (int b = 0; b < B_; ++b) {
      const float* zb = zl + b * ZDIM_ + j;
      acc[b] += h.x * zb[0] + h.y * zb[1] + h.z * zb[2] + h.w * zb[3];
    }
  }
#pragma unroll
  for (int b = 0; b < B_; ++b) delta[o * B_ + b] = acc[b];
}

// ---------------- kernel 2: standardize + pack weights to f16 ------------------
// one block (128 thr) per (b, oc); writes wt[b][r][oc][ic], r = kh*3+kw
__global__ __launch_bounds__(128) void standardize_kernel(const float* __restrict__ delta,
                                                          const float* __restrict__ base,
                                                          _Float16* __restrict__ wt) {
  const int b  = blockIdx.x >> 8;
  const int oc = blockIdx.x & 255;
  const int t  = threadIdx.x;

  float dv[9];
  float s = 0.f, s2 = 0.f;
#pragma unroll
  for (int k = 0; k < 9; ++k) {
    const int f = t + k * 128;
    float d = delta[((size_t)oc * FANIN_ + f) * B_ + b];
    dv[k] = d; s += d; s2 += d * d;
  }
#pragma unroll
  for (int m = 16; m; m >>= 1) { s += __shfl_xor(s, m, 32); s2 += __shfl_xor(s2, m, 32); }

  __shared__ float rs[4], rs2[4];
  if ((t & 31) == 0) { rs[t >> 5] = s; rs2[t >> 5] = s2; }
  __syncthreads();
  s  = rs[0]  + rs[1]  + rs[2]  + rs[3];
  s2 = rs2[0] + rs2[1] + rs2[2] + rs2[3];

  const float mu  = s * (1.0f / FANIN_);
  const float var = s2 * (1.0f / FANIN_) - mu * mu;
  const float scale = rsqrtf(var + 1e-5f) * 0.041666667f;   // sqrt(2/1152)
  const float inv_sqrt2 = 0.70710678f;

#pragma unroll
  for (int k = 0; k < 9; ++k) {
    const int f = t + k * 128;
    float w = (base[(size_t)oc * FANIN_ + f] + (dv[k] - mu) * scale) * inv_sqrt2;
    const int ic = f / 9, r = f - ic * 9;
    wt[(((size_t)b * 9 + r) * OUTC_ + oc) * INC_ + ic] = (_Float16)w;
  }
}

// ---------------- async global->LDS copy (CDNA5 ASYNCcnt path) -----------------
__device__ __forceinline__ void async_copy_b128(const _Float16* gsrc, _Float16* ldst) {
  // LDS byte address = low 32 bits of the generic pointer (ISA: addr[31:0])
  unsigned lds_addr = (unsigned)(size_t)ldst;
  asm volatile("global_load_async_to_lds_b128 %0, %1, off"
               :: "v"(lds_addr), "v"(gsrc) : "memory");
}
__device__ __forceinline__ void wait_async0() {
  asm volatile("s_wait_asynccnt 0" ::: "memory");
}

// ---------------- kernel 3: batched implicit-GEMM conv via WMMA f16 ------------
// grid (32 Ntiles, 2 Mtiles, 16 batch), 256 threads = 8 waves.
// Block tile: 128 M x 128 N; K = 9 taps x 128 ic, stepped 32 at a time.
// Double-buffered LDS; A-tile staged with global_load_async_to_lds_b128.
__global__ __launch_bounds__(256) void conv_wmma_kernel(const _Float16* __restrict__ xh,
                                                        const _Float16* __restrict__ wt,
                                                        float* __restrict__ out) {
  const int nt = blockIdx.x, mt = blockIdx.y, b = blockIdx.z;
  const int t = threadIdx.x;
  const int lane = t & 31, wv = t >> 5;

  __shared__ __align__(32) _Float16 lA[2][128 * 32];   // [m][k]   2 x 8 KB
  __shared__ __align__(32) _Float16 lB[2][128 * 32];   // [n][k]T  2 x 8 KB

  v8f acc[8] = {};

  const int h0 = nt * 2;                 // N tile = two image rows
  // A loader: thread -> (row, half-row of 16)
  const int ar = t >> 1;
  const int ac = (t & 1) << 4;
  // B loader: thread -> (ic-in-block, 16-wide n segment)
  const int bi  = t >> 3;                // 0..31
  const int bs  = t & 7;                 // 0..7
  const int bh  = h0 + (bs >> 2);        // image row for this segment
  const int bw0 = (bs & 3) << 4;         // starting image col

  const int am = (wv << 4) + (lane & 15);      // A fragment row in tile
  const int kb = (lane >> 4) << 4;             // K half per lane group

  // stage one k-step (tap r2, ic-block icb2) into buffer bufi
  auto stage = [&](int r2, int icb2, int bufi) {
    // A: weights [128 oc][32 ic] -- two async b128 (2 x 16 B) per thread
    const _Float16* wsrc = wt + (((size_t)b * 9 + r2) * OUTC_ + mt * 128) * INC_;
    async_copy_b128(&wsrc[ar * INC_ + icb2 + ac], &lA[bufi][ar * 32 + ac]);
    async_copy_b128(&wsrc[ar * INC_ + icb2 + ac + 8], &lA[bufi][ar * 32 + ac + 8]);
    // B transposed with zero-padding: lB[n][k] = x[b][icb2+bi][bh+dh][w+dw]
    const int dh = r2 / 3 - 1, dw = r2 % 3 - 1;
    const int hh = bh + dh;
    const bool hok = (unsigned)hh < (unsigned)HW_;
    const _Float16* xsrc = xh + (((size_t)b * INC_ + icb2 + bi) * HW_ + hh) * HW_;
#pragma unroll
    for (int q = 0; q < 16; ++q) {
      const int wc = bw0 + q + dw;
      _Float16 val = (hok && (unsigned)wc < (unsigned)HW_) ? xsrc[wc] : (_Float16)0.f;
      lB[bufi][((bs << 4) + q) * 32 + bi] = val;
    }
  };

  stage(0, 0, 0);                        // prologue: fill buffer 0

  for (int s = 0; s < KSTEPS_; ++s) {
    const int buf = s & 1;
    wait_async0();                       // this wave's async LDS deposits done
    __syncthreads();                     // all waves' deposits visible

    if (s + 1 < KSTEPS_) {               // stage next step into other buffer
      const int r2 = (s + 1) >> 2, icb2 = ((s + 1) & 3) << 5;
      stage(r2, icb2, buf ^ 1);
    }
    if (s + 2 < KSTEPS_) {               // prefetch weight stream 2 steps ahead
      const int r3 = (s + 2) >> 2, icb3 = ((s + 2) & 3) << 5;
      __builtin_prefetch(wt + (((size_t)b * 9 + r3) * OUTC_ + mt * 128) * INC_
                            + ar * INC_ + icb3 + ac, 0, 1);
    }

    v16h afr = *(const v16h*)&lA[buf][am * 32 + kb];
#pragma unroll
    for (int j = 0; j < 8; ++j) {
      v16h bfr = *(const v16h*)&lB[buf][((j << 4) + (lane & 15)) * 32 + kb];
      acc[j] = __builtin_amdgcn_wmma_f32_16x16x32_f16(
          false, afr, false, bfr, (short)0, acc[j], false, false);
    }
  }

  // epilogue: C/D layout -> lanes 0-15: M=v, lanes 16-31: M=v+8, N=lane&15
  const int mrow0 = mt * 128 + (wv << 4) + ((lane >> 4) << 3);
  const int col0  = nt * 128 + (lane & 15);
#pragma unroll
  for (int j = 0; j < 8; ++j) {
#pragma unroll
    for (int v = 0; v < 8; ++v) {
      out[((size_t)b * OUTC_ + mrow0 + v) * NOUT_ + col0 + (j << 4)] = acc[j][v];
    }
  }
}

// ---------------- launcher -----------------------------------------------------
extern "C" void kernel_launch(void* const* d_in, const int* in_sizes, int n_in,
                              void* d_out, int out_size, void* d_ws, size_t ws_size,
                              hipStream_t stream) {
  const float* x    = (const float*)d_in[0];
  const float* z    = (const float*)d_in[1];
  const float* base = (const float*)d_in[2];
  const float* hw   = (const float*)d_in[3];
  float* out        = (float*)d_out;

  const size_t XH_BYTES = (size_t)B_ * INC_ * HW_ * HW_ * sizeof(_Float16);   // 16.8 MB
  const size_t WT_BYTES = (size_t)B_ * 9 * OUTC_ * INC_ * sizeof(_Float16);   //  9.4 MB
  _Float16* xh   = (_Float16*)d_ws;
  _Float16* wtp  = (_Float16*)((char*)d_ws + XH_BYTES);
  float*    delt = (float*)((char*)d_ws + XH_BYTES + WT_BYTES);               // 18.9 MB

  // 0) x -> f16 (8.4M elems, 4 per thread)
  cvt_x_kernel<<<(B_ * INC_ * HW_ * HW_) / 1024, 256, 0, stream>>>(x, xh);
  // 1) delta = z @ head_w^T  (HBM-bound: streams 302 MB of head_w once)
  delta_gemm_kernel<<<OTOT_ / 256, 256, 0, stream>>>(z, hw, delt);
  // 2) standardize per (b, oc) and pack per-sample f16 weights [b][tap][oc][ic]
  standardize_kernel<<<B_ * OUTC_, 128, 0, stream>>>(delt, base, wtp);
  // 3) 16 implicit GEMMs via v_wmma_f32_16x16x32_f16, double-buffered + async LDS
  conv_wmma_kernel<<<dim3(NOUT_ / 128, OUTC_ / 128, B_), 256, 0, stream>>>(xh, wtp, out);
}
